// Attention_51685636440596
// MI455X (gfx1250) — matrix operand order repaired and checked by reading, not compiled, not run on hardware
//
#include <hip/hip_runtime.h>
#include <math.h>

#define ATTN_DIM_C 2048
#define NUM_HEADS_C 16
#define HEAD_DIM_C 128
#define B_C 2
#define T_C 2048
#define MTOT_C (B_C * T_C) // 4096

typedef __attribute__((ext_vector_type(16))) _Float16 v16h;
typedef __attribute__((ext_vector_type(8)))  float    v8f;

union Frag { v16h v; uint4 q[2]; };

// Async global->LDS copy of 16 bytes per lane (CDNA5 GLOBAL_LOAD_ASYNC_TO_LDS_B128,
// tracked by ASYNCcnt; no VGPR round-trip, no loadcnt stall).
__device__ __forceinline__ void async_copy16(const _Float16* gsrc, _Float16* ldst) {
  unsigned lds_addr = (unsigned)(size_t)ldst;            // low 32 bits = LDS address
  unsigned long long gaddr = (unsigned long long)(size_t)gsrc;
  asm volatile("global_load_async_to_lds_b128 %0, %1, off"
               :: "v"(lds_addr), "v"(gaddr) : "memory");
}
__device__ __forceinline__ void wait_async0() {
  asm volatile("s_wait_asynccnt 0x0" ::: "memory");
}

// ---------------------------------------------------------------- cast f32->f16
__global__ void k_cast_f32_f16(const float* __restrict__ s,
                               _Float16* __restrict__ d, size_t n) {
  size_t i = (size_t)blockIdx.x * blockDim.x + threadIdx.x;
  if (i < n) d[i] = (_Float16)s[i];
}

// ---------------------------------------------------------------- RoPE + pack
// src: f32 [B*T, 2048]. dqk (optional): f16 [B,H,T,128] with RoPE applied and
// scaled by qscale. dv (optional): f16 [B,H,128,T] (transposed, no RoPE).
__global__ void k_rope_pack(const float* __restrict__ src,
                            const float* __restrict__ cosb,
                            const float* __restrict__ sinb,
                            _Float16* __restrict__ dqk,
                            _Float16* __restrict__ dv,
                            float qscale) {
  size_t idx = (size_t)blockIdx.x * blockDim.x + threadIdx.x;
  size_t total = (size_t)B_C * T_C * NUM_HEADS_C * 64;
  if (idx >= total) return;
  int d = (int)(idx & 63);
  size_t r = idx >> 6;
  int h = (int)(r % NUM_HEADS_C);
  size_t bt = r / NUM_HEADS_C; // b*T + t
  int t = (int)(bt % T_C);
  int b = (int)(bt / T_C);
  const float* sp = src + bt * ATTN_DIM_C + (size_t)h * HEAD_DIM_C;
  float x1 = sp[d], x2 = sp[d + 64];
  if (dqk) {
    const float* cp = cosb + bt * HEAD_DIM_C;
    const float* ip = sinb + bt * HEAD_DIM_C;
    float y1 = (x1 * cp[d]      - x2 * ip[d])      * qscale;
    float y2 = (x2 * cp[d + 64] + x1 * ip[d + 64]) * qscale;
    _Float16* dp = dqk + (((size_t)b * NUM_HEADS_C + h) * T_C + t) * HEAD_DIM_C;
    dp[d]      = (_Float16)y1;
    dp[d + 64] = (_Float16)y2;
  }
  if (dv) {
    _Float16* vp = dv + ((size_t)b * NUM_HEADS_C + h) * HEAD_DIM_C * (size_t)T_C;
    vp[(size_t)d * T_C + t]        = (_Float16)x1;
    vp[(size_t)(d + 64) * T_C + t] = (_Float16)x2;
  }
}

// ---------------------------------------------------------------- f16 GEMM
// C[M,N] (f32) = A[M,K] (f16) * W[N,K]^T (f16). 128x128 block tile, K-step 32.
// Double-buffered LDS tiles filled with GLOBAL_LOAD_ASYNC_TO_LDS_B128 so the
// copy of step n+1 overlaps the WMMAs of step n.
__global__ __launch_bounds__(256)
void k_gemm_f16(const _Float16* __restrict__ A,
                const _Float16* __restrict__ W,
                float* __restrict__ C,
                int M, int N, int K) {
  __shared__ alignas(16) _Float16 sA[2][128][32];
  __shared__ alignas(16) _Float16 sB[2][128][32];
  const int tid  = threadIdx.x;
  const int lane = tid & 31;
  const int wave = tid >> 5;
  const int wm   = wave >> 2;
  const int wn   = wave & 3;
  const int bm   = blockIdx.x * 128;
  const int bn   = blockIdx.y * 128;
  const int row16 = lane & 15;
  const int kbA = (lane < 16) ? 0 : 8;   // A frag: chunks at kbA and 16+kbA
  const int kbB = (lane < 16) ? 0 : 8;   // B frag: chunks at kbB and 16+kbB

  // this thread's two 16B chunks of each tile
  const int c0  = tid * 2;
  const int tr0 = c0 >> 2,       tc0 = (c0 & 3) * 8;
  const int tr1 = (c0 + 1) >> 2, tc1 = ((c0 + 1) & 3) * 8;

  v8f zero8 = {0.f, 0.f, 0.f, 0.f, 0.f, 0.f, 0.f, 0.f};
  v8f acc[4][2];
  #pragma unroll
  for (int mi = 0; mi < 4; ++mi)
    #pragma unroll
    for (int ni = 0; ni < 2; ++ni) acc[mi][ni] = zero8;

  // prologue: async-fill buffer 0
  async_copy16(&A[(size_t)(bm + tr0) * K + tc0], &sA[0][tr0][tc0]);
  async_copy16(&A[(size_t)(bm + tr1) * K + tc1], &sA[0][tr1][tc1]);
  async_copy16(&W[(size_t)(bn + tr0) * K + tc0], &sB[0][tr0][tc0]);
  async_copy16(&W[(size_t)(bn + tr1) * K + tc1], &sB[0][tr1][tc1]);

  for (int k0 = 0; k0 < K; k0 += 32) {
    const int cur = (k0 >> 5) & 1;
    wait_async0();
    __syncthreads();                       // buffer `cur` complete for all waves
    if (k0 + 32 < K) {                     // prefetch next step into other buffer
      const int nxt = cur ^ 1;
      const int kn  = k0 + 32;
      async_copy16(&A[(size_t)(bm + tr0) * K + kn + tc0], &sA[nxt][tr0][tc0]);
      async_copy16(&A[(size_t)(bm + tr1) * K + kn + tc1], &sA[nxt][tr1][tc1]);
      async_copy16(&W[(size_t)(bn + tr0) * K + kn + tc0], &sB[nxt][tr0][tc0]);
      async_copy16(&W[(size_t)(bn + tr1) * K + kn + tc1], &sB[nxt][tr1][tc1]);
    }
    Frag af[4], bf[2];
    #pragma unroll
    for (int mi = 0; mi < 4; ++mi) {
      int rr = wm * 64 + mi * 16 + row16;
      af[mi].q[0] = *reinterpret_cast<const uint4*>(&sA[cur][rr][kbA]);
      af[mi].q[1] = *reinterpret_cast<const uint4*>(&sA[cur][rr][16 + kbA]);
    }
    #pragma unroll
    for (int ni = 0; ni < 2; ++ni) {
      int rr = wn * 32 + ni * 16 + row16;
      bf[ni].q[0] = *reinterpret_cast<const uint4*>(&sB[cur][rr][kbB]);
      bf[ni].q[1] = *reinterpret_cast<const uint4*>(&sB[cur][rr][kbB + 16]);
    }
    #pragma unroll
    for (int mi = 0; mi < 4; ++mi)
      #pragma unroll
      for (int ni = 0; ni < 2; ++ni)
        acc[mi][ni] = __builtin_amdgcn_wmma_f32_16x16x32_f16(
            false, af[mi].v, false, bf[ni].v, (short)0, acc[mi][ni], false, false);
    __syncthreads();                       // all reads of `cur` done before reuse
  }
  const int rowoff = (lane < 16) ? 0 : 8;
  #pragma unroll
  for (int mi = 0; mi < 4; ++mi)
    #pragma unroll
    for (int ni = 0; ni < 2; ++ni)
      #pragma unroll
      for (int j = 0; j < 8; ++j) {
        int rr = bm + wm * 64 + mi * 16 + j + rowoff;
        int cc = bn + wn * 32 + ni * 16 + row16;
        C[(size_t)rr * N + cc] = acc[mi][ni][j];
      }
}

// ---------------------------------------------------------------- flash attention
// Q,K: f16 [B,H,T,128] (Q pre-scaled by 1/sqrt(128)); V: f16 [B,H,128,T].
// O: f32 [B*T, 2048]. One wave owns 16 query rows; 32-key blocks, online softmax.
__global__ __launch_bounds__(256)
void k_attn(const _Float16* __restrict__ Q,
            const _Float16* __restrict__ Kt,
            const _Float16* __restrict__ Vt,
            float* __restrict__ O) {
  __shared__ alignas(16) _Float16 pshm[8][512]; // per-wave 16x32 P tile
  const int tid  = threadIdx.x;
  const int lane = tid & 31;
  const int wave = tid >> 5;
  const int gw   = blockIdx.x * 8 + wave;   // 0 .. B*H*(T/16)-1
  const int tilesPerHead = T_C / 16;
  const int qt = (gw % tilesPerHead) * 16;
  const int bh = gw / tilesPerHead;         // b*H + h
  const int b  = bh / NUM_HEADS_C;
  const int h  = bh % NUM_HEADS_C;

  const int row16 = lane & 15;
  const int kbA = (lane < 16) ? 0 : 8;
  const int kbB = (lane < 16) ? 0 : 8;
  const int rowoff = (lane < 16) ? 0 : 8;

  const _Float16* qp = Q + ((size_t)bh * T_C + qt + row16) * HEAD_DIM_C;
  Frag qf[4];
  #pragma unroll
  for (int kk = 0; kk < 4; ++kk) {
    qf[kk].q[0] = *reinterpret_cast<const uint4*>(&qp[kk * 32 + kbA]);
    qf[kk].q[1] = *reinterpret_cast<const uint4*>(&qp[kk * 32 + 16 + kbA]);
  }

  v8f zero8 = {0.f, 0.f, 0.f, 0.f, 0.f, 0.f, 0.f, 0.f};
  float mstat[8], lstat[8];
  v8f acc[8];
  #pragma unroll
  for (int j = 0; j < 8; ++j) { mstat[j] = -3.0e38f; lstat[j] = 0.f; }
  #pragma unroll
  for (int n = 0; n < 8; ++n) acc[n] = zero8;

  const _Float16* kbase = Kt + (size_t)bh * T_C * HEAD_DIM_C;
  const _Float16* vbase = Vt + (size_t)bh * HEAD_DIM_C * (size_t)T_C;

  for (int s0 = 0; s0 < qt + 16; s0 += 32) {
    // prefetch next key/value block while we compute this one
    if (s0 + 32 < qt + 16) {
      __builtin_prefetch(kbase + (size_t)(s0 + 32 + row16) * HEAD_DIM_C, 0, 0);
      __builtin_prefetch(vbase + (size_t)row16 * T_C + s0 + 32, 0, 0);
    }
    // S = Q * K^T for this 16x32 block (two 16x16 C tiles, K=128)
    v8f sv[2];
    sv[0] = zero8; sv[1] = zero8;
    #pragma unroll
    for (int nt = 0; nt < 2; ++nt) {
      const _Float16* kp = kbase + (size_t)(s0 + nt * 16 + row16) * HEAD_DIM_C;
      #pragma unroll
      for (int kk = 0; kk < 4; ++kk) {
        Frag kf;
        kf.q[0] = *reinterpret_cast<const uint4*>(&kp[kk * 32 + kbB]);
        kf.q[1] = *reinterpret_cast<const uint4*>(&kp[kk * 32 + kbB + 16]);
        sv[nt] = __builtin_amdgcn_wmma_f32_16x16x32_f16(
            false, qf[kk].v, false, kf.v, (short)0, sv[nt], false, false);
      }
    }
    // causal mask + online softmax (rows split across lane halves)
    #pragma unroll
    for (int j = 0; j < 8; ++j) {
      int trow = qt + j + rowoff;
      #pragma unroll
      for (int nt = 0; nt < 2; ++nt) {
        int sc = s0 + nt * 16 + row16;
        if (sc > trow) sv[nt][j] = -10000.0f;
      }
      float mx = fmaxf(sv[0][j], sv[1][j]);
      #pragma unroll
      for (int off = 1; off < 16; off <<= 1)
        mx = fmaxf(mx, __shfl_xor(mx, off, 32));
      float mnew  = fmaxf(mstat[j], mx);
      float scale = __expf(mstat[j] - mnew);
      float p0 = __expf(sv[0][j] - mnew);
      float p1 = __expf(sv[1][j] - mnew);
      float rs = p0 + p1;
      #pragma unroll
      for (int off = 1; off < 16; off <<= 1)
        rs += __shfl_xor(rs, off, 32);
      lstat[j] = lstat[j] * scale + rs;
      mstat[j] = mnew;
      #pragma unroll
      for (int n = 0; n < 8; ++n) acc[n][j] *= scale;
      // stash P (C-layout -> LDS row-major 16x32)
      pshm[wave][(j + rowoff) * 32 + row16]      = (_Float16)p0;
      pshm[wave][(j + rowoff) * 32 + 16 + row16] = (_Float16)p1;
    }
    // LDS ops are in-order within a wave: reload P in A-fragment layout.
    Frag pf;
    const _Float16* pp = &pshm[wave][row16 * 32];
    pf.q[0] = *reinterpret_cast<const uint4*>(&pp[kbA]);
    pf.q[1] = *reinterpret_cast<const uint4*>(&pp[16 + kbA]);
    #pragma unroll
    for (int n = 0; n < 8; ++n) {
      Frag vf;
      const _Float16* vp = vbase + (size_t)(n * 16 + row16) * T_C + s0;
      vf.q[0] = *reinterpret_cast<const uint4*>(&vp[kbB]);
      vf.q[1] = *reinterpret_cast<const uint4*>(&vp[kbB + 16]);
      acc[n] = __builtin_amdgcn_wmma_f32_16x16x32_f16(
          false, pf.v, false, vf.v, (short)0, acc[n], false, false);
    }
  }

  #pragma unroll
  for (int n = 0; n < 8; ++n)
    #pragma unroll
    for (int j = 0; j < 8; ++j) {
      int t = qt + j + rowoff;
      int d = n * 16 + row16;
      O[((size_t)b * T_C + t) * ATTN_DIM_C + (size_t)h * HEAD_DIM_C + d] =
          acc[n][j] / lstat[j];
    }
}

// ---------------------------------------------------------------- launch
extern "C" void kernel_launch(void* const* d_in, const int* in_sizes, int n_in,
                              void* d_out, int out_size, void* d_ws, size_t ws_size,
                              hipStream_t stream) {
  (void)in_sizes; (void)n_in; (void)out_size; (void)ws_size;
  const float* x    = (const float*)d_in[0];
  const float* cosb = (const float*)d_in[1];
  const float* sinb = (const float*)d_in[2];
  const float* wq   = (const float*)d_in[3];
  const float* wk   = (const float*)d_in[4];
  const float* wv   = (const float*)d_in[5];
  const float* wo   = (const float*)d_in[6];

  const size_t MB = 1ull << 20;   // workspace plan: 144 MB total
  char* ws = (char*)d_ws;
  _Float16* xh  = (_Float16*)(ws + 0);        // 16 MB
  _Float16* wqh = (_Float16*)(ws + 16 * MB);  //  8 MB each
  _Float16* wkh = (_Float16*)(ws + 24 * MB);
  _Float16* wvh = (_Float16*)(ws + 32 * MB);
  _Float16* woh = (_Float16*)(ws + 40 * MB);
  float*    sf  = (float*)   (ws + 48 * MB);  // 32 MB f32 staging (q/k/v/o)
  _Float16* qh  = (_Float16*)(ws + 80 * MB);  // 16 MB
  _Float16* kh  = (_Float16*)(ws + 96 * MB);  // 16 MB
  _Float16* vh  = (_Float16*)(ws + 112 * MB); // 16 MB
  _Float16* oh  = (_Float16*)(ws + 128 * MB); // 16 MB

  const size_t nX = (size_t)MTOT_C * ATTN_DIM_C;
  const size_t nW = (size_t)ATTN_DIM_C * ATTN_DIM_C;

  k_cast_f32_f16<<<(unsigned)(nX / 256), 256, 0, stream>>>(x,  xh,  nX);
  k_cast_f32_f16<<<(unsigned)(nW / 256), 256, 0, stream>>>(wq, wqh, nW);
  k_cast_f32_f16<<<(unsigned)(nW / 256), 256, 0, stream>>>(wk, wkh, nW);
  k_cast_f32_f16<<<(unsigned)(nW / 256), 256, 0, stream>>>(wv, wvh, nW);
  k_cast_f32_f16<<<(unsigned)(nW / 256), 256, 0, stream>>>(wo, woh, nW);

  dim3 gg(MTOT_C / 128, ATTN_DIM_C / 128);
  const unsigned ropeBlocks =
      (unsigned)(((size_t)B_C * T_C * NUM_HEADS_C * 64) / 256);
  const float qs = 1.0f / sqrtf((float)HEAD_DIM_C);

  k_gemm_f16<<<gg, 256, 0, stream>>>(xh, wqh, sf, MTOT_C, ATTN_DIM_C, ATTN_DIM_C);
  k_rope_pack<<<ropeBlocks, 256, 0, stream>>>(sf, cosb, sinb, qh, nullptr, qs);
  k_gemm_f16<<<gg, 256, 0, stream>>>(xh, wkh, sf, MTOT_C, ATTN_DIM_C, ATTN_DIM_C);
  k_rope_pack<<<ropeBlocks, 256, 0, stream>>>(sf, cosb, sinb, kh, nullptr, 1.0f);
  k_gemm_f16<<<gg, 256, 0, stream>>>(xh, wvh, sf, MTOT_C, ATTN_DIM_C, ATTN_DIM_C);
  k_rope_pack<<<ropeBlocks, 256, 0, stream>>>(sf, nullptr, nullptr, nullptr, vh, 0.f);

  k_attn<<<(unsigned)(B_C * NUM_HEADS_C * (T_C / 16) / 8), 256, 0, stream>>>(
      qh, kh, vh, sf);
  k_cast_f32_f16<<<(unsigned)(nX / 256), 256, 0, stream>>>(sf, oh, nX);
  k_gemm_f16<<<gg, 256, 0, stream>>>(oh, woh, (float*)d_out,
                                     MTOT_C, ATTN_DIM_C, ATTN_DIM_C);
}
